// TLSTM_30657476559121
// MI455X (gfx1250) — matrix operand order repaired and checked by reading, not compiled, not run on hardware
//
#include <hip/hip_runtime.h>

// ---------------------------------------------------------------------------
// T-LSTM (bidirectional) for MI455X / gfx1250.
//   Phase 1: XW = bf16(X) @ bf16([Wi|Wf|Wog|Wc])  via v_wmma_f32_16x16x32_bf16,
//            stored bf16 (32MB, L2-resident).
//   Phase 2: persistent per-direction scan kernel, h/c state in LDS,
//            per-step h@U / c@Wd GEMMs via WMMA; per-step XW gate tile
//            (32x1024 bf16, row stride S*NG) prefetched by the Tensor Data
//            Mover into a double-buffered LDS tile (tensor_load_to_lds +
//            s_wait_tensorcnt), overlapping DMA with WMMA compute.
// Whole working set (~120MB) fits in the 192MB L2 -> per-step U re-reads and
// XW tiles are L2 hits; HBM is touched essentially once.
// ---------------------------------------------------------------------------

typedef unsigned int u32;
typedef unsigned long long u64;
typedef __attribute__((ext_vector_type(4)))  u32    u32x4;
typedef __attribute__((ext_vector_type(4)))  int    i32x4;
typedef __attribute__((ext_vector_type(8)))  int    i32x8;
typedef __attribute__((ext_vector_type(16))) __bf16 v16bf;
typedef __attribute__((ext_vector_type(8)))  float  v8f;

union FragBF { u32x4 u4[2]; v16bf v; };

#define NB   32      // batch
#define NS   512     // sequence
#define NI   768     // input dim
#define NH   256     // hidden dim
#define NG   1024    // 4*NH gate columns
#define NROW (NB*NS) // 16384 rows of X

#if defined(__has_builtin)
#if __has_builtin(__builtin_amdgcn_tensor_load_to_lds) && __has_builtin(__builtin_amdgcn_s_wait_tensorcnt)
#define USE_TDM 1
#endif
#endif
#ifndef USE_TDM
#define USE_TDM 0
#endif

__device__ __forceinline__ unsigned short f2bf(float x) {
  union { float f; u32 u; } c; c.f = x;
  u32 u = c.u;
  return (unsigned short)((u + 0x7FFFu + ((u >> 16) & 1u)) >> 16); // RNE
}
__device__ __forceinline__ float bf2f(unsigned short h) {
  union { u32 u; float f; } c; c.u = ((u32)h) << 16;
  return c.f;
}

// A fragment (16x32 bf16, M x K) from row-major [*, ld] bf16 matrix.
// ISA layout: lane<16 -> M=lane, K in {0..7,16..23}; lane>=16 -> K in {8..15,24..31}.
__device__ __forceinline__ v16bf load_fragA(const unsigned short* __restrict__ base,
                                            int row, int ld, int k0, int lane) {
  int kb = k0 + ((lane >> 4) << 3);
  FragBF f;
  f.u4[0] = *(const u32x4*)(base + (size_t)row * ld + kb);
  f.u4[1] = *(const u32x4*)(base + (size_t)row * ld + kb + 16);
  return f.v;
}

// B fragment (32x16 bf16, K x N) from N-major [N][K] bf16 matrix (i.e. U^T).
// ISA layout: lane<16 -> N=lane, K in {0..7,16..23}; lane>=16 -> N=lane-16, K in {8..15,24..31}.
__device__ __forceinline__ v16bf load_fragB(const unsigned short* __restrict__ base,
                                            int n, int ld, int k0, int lane) {
  int kk = k0 + ((lane >> 4) << 3);
  FragBF f;
  f.u4[0] = *(const u32x4*)(base + (size_t)n * ld + kk);
  f.u4[1] = *(const u32x4*)(base + (size_t)n * ld + kk + 16);
  return f.v;
}

// -------------------------- pack / convert kernels --------------------------

__global__ void cvt_x(const float* __restrict__ X, unsigned short* __restrict__ Xbf, int n) {
  int i = blockIdx.x * blockDim.x + threadIdx.x;
  if (i < n) Xbf[i] = f2bf(X[i]);
}

// WT[(g*256+h)][k] = Wg[k][h]   (1024 x 768, bf16, N-major)
__global__ void pack_w(const float* __restrict__ Wi, const float* __restrict__ Wf,
                       const float* __restrict__ Wog, const float* __restrict__ Wc,
                       unsigned short* __restrict__ WT) {
  int i = blockIdx.x * blockDim.x + threadIdx.x;
  if (i >= NG * NI) return;
  int n = i / NI, k = i % NI;
  int g = n >> 8, h = n & 255;
  const float* W = (g == 0) ? Wi : (g == 1) ? Wf : (g == 2) ? Wog : Wc;
  WT[i] = f2bf(W[k * NH + h]);
}

// UT[n][k]: n<1024 -> U_g[k][h], n in [1024,1280) -> W_decomp[k][n-1024]
__global__ void pack_u(const float* __restrict__ Ui, const float* __restrict__ Uf,
                       const float* __restrict__ Uog, const float* __restrict__ Uc,
                       const float* __restrict__ Wd, unsigned short* __restrict__ UT) {
  int i = blockIdx.x * blockDim.x + threadIdx.x;
  if (i >= 1280 * NH) return;
  int n = i / NH, k = i % NH;
  const float* U;
  int h;
  if (n < NG) { int g = n >> 8; U = (g == 0) ? Ui : (g == 1) ? Uf : (g == 2) ? Uog : Uc; h = n & 255; }
  else        { U = Wd; h = n - NG; }
  UT[i] = f2bf(U[k * NH + h]);
}

__global__ void pack_b(const float* __restrict__ bi, const float* __restrict__ bfv,
                       const float* __restrict__ bog, const float* __restrict__ bc,
                       float* __restrict__ bpk) {
  int i = blockIdx.x * blockDim.x + threadIdx.x;
  if (i >= NG) return;
  int g = i >> 8, h = i & 255;
  bpk[i] = ((g == 0) ? bi : (g == 1) ? bfv : (g == 2) ? bog : bc)[h];
}

// ------------------------------ big GEMM: X@W -------------------------------
// Each wave: 32 rows x 64 cols (2x4 WMMA tiles), K-loop over 768; bf16 output.
__global__ void __launch_bounds__(256) gemm_xw(const unsigned short* __restrict__ Xbf,
                                               const unsigned short* __restrict__ WT,
                                               unsigned short* __restrict__ XWbf) {
  int wid  = (blockIdx.x * blockDim.x + threadIdx.x) >> 5;
  int lane = threadIdx.x & 31;
  int ngrp = wid & 15;   // 16 groups of 64 cols
  int mgrp = wid >> 4;   // 512 groups of 32 rows
  int laneN = lane & 15;
  int mOff  = (lane >> 4) << 3;

  v8f acc[2][4];
#pragma unroll
  for (int mt = 0; mt < 2; ++mt)
#pragma unroll
    for (int nt = 0; nt < 4; ++nt) acc[mt][nt] = (v8f)0.0f;

  for (int ks = 0; ks < NI; ks += 32) {
    v16bf a0 = load_fragA(Xbf, mgrp * 32 + laneN,      NI, ks, lane);
    v16bf a1 = load_fragA(Xbf, mgrp * 32 + 16 + laneN, NI, ks, lane);
#pragma unroll
    for (int nt = 0; nt < 4; ++nt) {
      v16bf b = load_fragB(WT, ngrp * 64 + nt * 16 + laneN, NI, ks, lane);
      acc[0][nt] = __builtin_amdgcn_wmma_f32_16x16x32_bf16(false, a0, false, b, (short)0, acc[0][nt], false, false);
      acc[1][nt] = __builtin_amdgcn_wmma_f32_16x16x32_bf16(false, a1, false, b, (short)0, acc[1][nt], false, false);
    }
  }
#pragma unroll
  for (int mt = 0; mt < 2; ++mt)
#pragma unroll
    for (int nt = 0; nt < 4; ++nt)
#pragma unroll
      for (int r = 0; r < 8; ++r) {
        int row = mgrp * 32 + mt * 16 + r + mOff;
        int col = ngrp * 64 + nt * 16 + laneN;
        XWbf[(size_t)row * NG + col] = f2bf(acc[mt][nt][r]);
      }
}

// ----------------------------- recurrent scan -------------------------------
// One workgroup per direction; 16 waves; per step 160 WMMA C-tiles.
// XW gate tile for step k+1 is DMA'd by the TDM into LDS while step k computes.
__global__ void __launch_bounds__(512, 1) tlstm_scan(const unsigned short* __restrict__ XWbf,
                                                     const unsigned short* __restrict__ UT,
                                                     const float* __restrict__ bpk,
                                                     const float* __restrict__ bdec,
                                                     const float* __restrict__ times,
                                                     float* __restrict__ out) {
  __shared__ unsigned short hs_bf[NB][NH];     // 16 KB   h state (bf16, matmul A)
  __shared__ unsigned short cs_bf[NB][NH];     // 16 KB   c state (bf16, matmul A)
  __shared__ float          cs_f [NB][NH];     // 32 KB   c state (f32, exact)
  __shared__ unsigned short Gact [NB][NG];     // 64 KB   sigmoided gates
  __shared__ unsigned short CST  [NB][NH];     // 16 KB   tanh(c@Wd + bd)
#if USE_TDM
  __shared__ unsigned short XWbuf[2][NB][NG];  // 128 KB  double-buffered XW tile
#endif

  const int dir  = blockIdx.x;
  const int tid  = threadIdx.x;
  const int lane = tid & 31;
  const int wave = tid >> 5;
  const int laneN = lane & 15;
  const int mOff  = (lane >> 4) << 3;

#if USE_TDM
  // TDM D# descriptor (ISA 8.3-8.5): 2D tile, 32 rows x 1024 bf16 cols,
  // row stride NS*NG elements; dest = XWbuf[parity].
  auto tdm_issue = [&](int step, int parity) {
    int sx = dir ? (NS - 1 - step) : step;
    u64 ga = (u64)(uintptr_t)(XWbf + (size_t)sx * NG);      // byte addr of row b=0
    u32 lds = (u32)(uintptr_t)(&XWbuf[parity][0][0]);
    u32x4 g0;
    g0[0] = 1u;                                             // count=1, user desc
    g0[1] = lds;                                            // lds_addr
    g0[2] = (u32)ga;                                        // global_addr[31:0]
    g0[3] = (u32)((ga >> 32) & 0x1FFFFFFu) | (2u << 30);    // ga[56:32] | type=2
    i32x8 g1;
    g1[0] = (int)(1u << 16);                  // wg_mask=0, data_size=1 (2B)
    g1[1] = (int)((u32)(NG & 0xFFFF) << 16);  // tensor_dim0[15:0]
    g1[2] = (int)((u32)(NROW & 0xFFFF) << 16);// tensor_dim0[31:16]=0 | tensor_dim1[15:0]
    g1[3] = (int)((u32)NG << 16);             // tensor_dim1[31:16]=0 | tile_dim0=1024
    g1[4] = (int)NB;                          // tile_dim1=32, tile_dim2=0
    g1[5] = (int)(NS * NG);                   // tensor_dim0_stride[31:0]
    g1[6] = 0;                                // stride[47:32]=0, dim1_stride lo=0
    g1[7] = 0;
    i32x4 gz4 = (i32x4)0;                     // group2/group3: 2-D tensor
    i32x8 gz8 = (i32x8)0;                     // extra arg (clang-23 6-arg form)
    __builtin_amdgcn_tensor_load_to_lds(g0, g1, gz4, gz4, gz8, 0);
  };
#endif

  for (int i = tid; i < NB * NH; i += 512) {
    int b = i >> 8, h = i & 255;
    hs_bf[b][h] = 0; cs_bf[b][h] = 0; cs_f[b][h] = 0.0f;
  }
#if USE_TDM
  if (wave == 0) tdm_issue(0, 0);
#endif
  __syncthreads();

  for (int k = 0; k < NS; ++k) {
    const int sx = dir ? (NS - 1 - k) : k;                     // input row
    const int st = dir ? ((k == 0) ? 0 : (NS - k)) : k;        // time index

#if USE_TDM
    // Prefetch step k+1's XW tile, then wait for step k's tile (in-order TDM).
    if (wave == 0) {
      if (k + 1 < NS) {
        tdm_issue(k + 1, (k + 1) & 1);
        __builtin_amdgcn_s_wait_tensorcnt(1);   // step k's DMA (older) is done
      } else {
        __builtin_amdgcn_s_wait_tensorcnt(0);   // last step: drain
      }
    }
    __syncthreads();   // publish TDM-written LDS to all waves
#endif

    // ---- phase A: WMMA matmuls (gates: h@U, decomp: c@Wd) ----
    for (int j = 0; j < 5; ++j) {
      const int nt = wave * 5 + j;                 // 0..79 N-tiles
      const bool isGate = nt < 64;
      const int colbase = isGate ? nt * 16 : (nt - 64) * 16;
      const int urow    = nt * 16;                 // row in UT [1280][256]
      v8f acc[2];
#pragma unroll
      for (int mt = 0; mt < 2; ++mt)
#pragma unroll
        for (int r = 0; r < 8; ++r) {
          int b = mt * 16 + r + mOff;
          int col = colbase + laneN;
          float v;
          if (isGate) {
#if USE_TDM
            v = bf2f(XWbuf[k & 1][b][col]) + bpk[col];
#else
            v = bf2f(XWbf[((size_t)b * NS + sx) * NG + col]) + bpk[col];
#endif
          } else {
            v = bdec[col];
          }
          acc[mt][r] = v;
        }
#pragma unroll
      for (int ks = 0; ks < NH; ks += 32) {
        int kk = ks + ((lane >> 4) << 3);
        FragBF bfrag;
        bfrag.u4[0] = *(const u32x4*)(UT + (size_t)(urow + laneN) * NH + kk);
        bfrag.u4[1] = *(const u32x4*)(UT + (size_t)(urow + laneN) * NH + kk + 16);
        int kb = ks + mOff;  // (lane>>4)*8
#pragma unroll
        for (int mt = 0; mt < 2; ++mt) {
          int arow = mt * 16 + laneN;
          FragBF afrag;
          afrag.u4[0] = isGate ? *(const u32x4*)&hs_bf[arow][kb]
                               : *(const u32x4*)&cs_bf[arow][kb];
          afrag.u4[1] = isGate ? *(const u32x4*)&hs_bf[arow][kb + 16]
                               : *(const u32x4*)&cs_bf[arow][kb + 16];
          acc[mt] = __builtin_amdgcn_wmma_f32_16x16x32_bf16(false, afrag.v, false, bfrag.v,
                                                            (short)0, acc[mt], false, false);
        }
      }
#pragma unroll
      for (int mt = 0; mt < 2; ++mt)
#pragma unroll
        for (int r = 0; r < 8; ++r) {
          int b = mt * 16 + r + mOff;
          int col = colbase + laneN;
          float v = acc[mt][r];
          if (isGate) Gact[b][col] = f2bf(1.0f / (1.0f + __expf(-v)));   // sigmoid
          else        CST[b][col]  = f2bf(tanhf(v));
        }
    }
    __syncthreads();

    // ---- phase B: elementwise T-LSTM update ----
    {
      int b = tid >> 4;
      int hbase = (tid & 15) << 4;
      float t = times[(size_t)b * NS + st];
      float T = 1.0f / logf(t + 2.7182818284590452f);
#pragma unroll
      for (int hh = 0; hh < 16; ++hh) {
        int h = hbase + hh;
        float cst = bf2f(CST[b][h]);
        float csn = cs_f[b][h] - cst + T * cst;   // subspace decay
        float ig = bf2f(Gact[b][h]);
        float fg = bf2f(Gact[b][NH + h]);
        float og = bf2f(Gact[b][2 * NH + h]);
        float cc = bf2f(Gact[b][3 * NH + h]);     // reference: sigmoid candidate
        float ct = fg * csn + ig * cc;
        float ht = og * tanhf(ct);
        cs_f[b][h]  = ct;
        cs_bf[b][h] = f2bf(ct);
        hs_bf[b][h] = f2bf(ht);
        out[((size_t)b * NS + k) * (2 * NH) + dir * NH + h] = ht;
        if (k == NS - 1) {
          float* hl = out + (size_t)NB * NS * (2 * NH);
          float* cl = hl + (size_t)NB * (2 * NH);
          hl[b * (2 * NH) + dir * NH + h] = ht;
          cl[b * (2 * NH) + dir * NH + h] = ct;
        }
      }
    }
    __syncthreads();
  }
}

// ------------------------------- launcher -----------------------------------

extern "C" void kernel_launch(void* const* d_in, const int* in_sizes, int n_in,
                              void* d_out, int out_size, void* d_ws, size_t ws_size,
                              hipStream_t stream) {
  const float* inputs = (const float*)d_in[0];
  const float* times  = (const float*)d_in[1];
  const float* Wi  = (const float*)d_in[2];
  const float* Ui  = (const float*)d_in[3];
  const float* bi  = (const float*)d_in[4];
  const float* Wf  = (const float*)d_in[5];
  const float* Uf  = (const float*)d_in[6];
  const float* bfv = (const float*)d_in[7];
  const float* Wog = (const float*)d_in[8];
  const float* Uog = (const float*)d_in[9];
  const float* bog = (const float*)d_in[10];
  const float* Wc  = (const float*)d_in[11];
  const float* Uc  = (const float*)d_in[12];
  const float* bc  = (const float*)d_in[13];
  const float* Wd  = (const float*)d_in[14];
  const float* bd  = (const float*)d_in[15];
  float* out = (float*)d_out;

  char* p = (char*)d_ws;
  auto take = [&](size_t bytes) {
    char* r = p;
    p += (bytes + 255) & ~(size_t)255;
    return r;
  };
  unsigned short* Xbf  = (unsigned short*)take((size_t)NROW * NI * 2);   // 24 MB
  unsigned short* WT   = (unsigned short*)take((size_t)NG * NI * 2);     // 1.5 MB
  unsigned short* UT   = (unsigned short*)take((size_t)1280 * NH * 2);   // 640 KB
  float*          bpk  = (float*)take((size_t)NG * 4);                   // 4 KB
  unsigned short* XWbf = (unsigned short*)take((size_t)NROW * NG * 2);   // 32 MB

  const int nX = NROW * NI;
  cvt_x <<<(nX + 255) / 256, 256, 0, stream>>>(inputs, Xbf, nX);
  pack_w<<<(NG * NI + 255) / 256, 256, 0, stream>>>(Wi, Wf, Wog, Wc, WT);
  pack_u<<<(1280 * NH + 255) / 256, 256, 0, stream>>>(Ui, Uf, Uog, Uc, Wd, UT);
  pack_b<<<(NG + 255) / 256, 256, 0, stream>>>(bi, bfv, bog, bc, bpk);
  gemm_xw<<<1024, 256, 0, stream>>>(Xbf, WT, XWbf);          // 8192 waves
  tlstm_scan<<<2, 512, 0, stream>>>(XWbf, UT, bpk, bd, times, out);
}
